// Structure_Decoder_32590211842241
// MI455X (gfx1250) — matrix operand order repaired and checked by reading, not compiled, not run on hardware
//
#include <hip/hip_runtime.h>

#define N_NODES 16384
#define N_HID   128
#define N_EDGES (N_NODES * 32)

typedef __attribute__((ext_vector_type(16))) _Float16 v16h;
typedef __attribute__((ext_vector_type(8)))  float    v8f;
typedef __attribute__((ext_vector_type(4)))  int      v4i;

typedef __attribute__((address_space(1))) v4i gas_v4i;   // global int4
typedef __attribute__((address_space(3))) v4i las_v4i;   // LDS int4

union FragH { v16h v; uint4 q[2]; };

// ---- CDNA5 async global->LDS copy (falls back to sync copy if unavailable) --
#if __has_builtin(__builtin_amdgcn_global_load_async_to_lds_b128)
#define ASYNC_TILE_LOADS 1
#endif

__device__ __forceinline__ void copy16_g2l(void* lds, const void* g) {
#if defined(ASYNC_TILE_LOADS)
    // one VMEM instruction, no VGPR staging, tracked by ASYNCcnt
    __builtin_amdgcn_global_load_async_to_lds_b128(
        (gas_v4i*)(g),
        (las_v4i*)(lds),
        /*imm offset*/ 0, /*cpol*/ 0);
#else
    *(uint4*)lds = *(const uint4*)g;
#endif
}

__device__ __forceinline__ void async_copies_wait() {
#if defined(ASYNC_TILE_LOADS)
#if __has_builtin(__builtin_amdgcn_s_wait_asynccnt)
    __builtin_amdgcn_s_wait_asynccnt(0);
#else
    asm volatile("s_wait_asynccnt 0x0" ::: "memory");
#endif
#endif
}

// ---------------------------------------------------------------- degrees ---
__global__ void k_init_deg(float* __restrict__ deg) {
    int i = blockIdx.x * blockDim.x + threadIdx.x;
    if (i < N_NODES) deg[i] = 1.0f;            // self-loop contributes 1
}

__global__ void k_edge_deg(const int* __restrict__ ei, float* __restrict__ deg) {
    int e = blockIdx.x * blockDim.x + threadIdx.x;
    if (e < N_EDGES) {
        int c = ei[N_EDGES + e];               // target node
        unsafeAtomicAdd(&deg[c], 1.0f);
    }
}

__global__ void k_dinv(const float* __restrict__ deg, float* __restrict__ dinv) {
    int i = blockIdx.x * blockDim.x + threadIdx.x;
    if (i < N_NODES) {
        float d = deg[i];
        dinv[i] = (d > 0.0f) ? rsqrtf(d) : 0.0f;
    }
}

// ------------------------------------------------------------- h = x @ W ---
// 32 rows per 256-thread block; W (64KB) resident in LDS, fp32 accumulate.
__global__ __launch_bounds__(256) void k_xw(const float* __restrict__ x,
                                            const float* __restrict__ W,
                                            float* __restrict__ h) {
    __shared__ __align__(16) float Ws[N_HID * N_HID];
    for (int t = threadIdx.x; t < N_HID * N_HID / 4; t += 256)
        copy16_g2l(&((float4*)Ws)[t], &((const float4*)W)[t]);
    async_copies_wait();
    __syncthreads();

    const int r  = blockIdx.x * 32 + (threadIdx.x >> 3);
    const int cg = (threadIdx.x & 7) * 16;
    const float4* x4 = (const float4*)(x + (size_t)r * N_HID);

    float acc[16];
#pragma unroll
    for (int j = 0; j < 16; ++j) acc[j] = 0.0f;

    for (int k4 = 0; k4 < N_HID / 4; ++k4) {
        float4 xv = x4[k4];
        const float* w0 = &Ws[(k4 * 4 + 0) * N_HID + cg];
        const float* w1 = &Ws[(k4 * 4 + 1) * N_HID + cg];
        const float* w2 = &Ws[(k4 * 4 + 2) * N_HID + cg];
        const float* w3 = &Ws[(k4 * 4 + 3) * N_HID + cg];
#pragma unroll
        for (int j = 0; j < 16; ++j)
            acc[j] += xv.x * w0[j] + xv.y * w1[j] + xv.z * w2[j] + xv.w * w3[j];
    }
    float* o = h + (size_t)r * N_HID + cg;
#pragma unroll
    for (int j = 0; j < 16; ++j) o[j] = acc[j];
}

// ----------------------------------------------- self-loop init of agg ------
__global__ void k_self(const float* __restrict__ h, const float* __restrict__ dinv,
                       float* __restrict__ agg) {
    int t = blockIdx.x * blockDim.x + threadIdx.x;
    if (t < N_NODES * N_HID) {
        float d = dinv[t >> 7];
        agg[t] = d * d * h[t];
    }
}

// ------------------------------------------------- edge scatter-add ---------
// one wave32 per edge, 4 floats per lane (float4 gather + 4 f32 atomics)
__global__ __launch_bounds__(256) void k_scatter(const int* __restrict__ ei,
                                                 const float* __restrict__ h,
                                                 const float* __restrict__ dinv,
                                                 float* __restrict__ agg) {
    const int e    = blockIdx.x * 8 + (threadIdx.x >> 5);
    const int lane = threadIdx.x & 31;
    const int r = ei[e];
    const int c = ei[N_EDGES + e];
    const float w = dinv[r] * dinv[c];
    const float4 hv = ((const float4*)(h + (size_t)r * N_HID))[lane];
    float* dst = agg + (size_t)c * N_HID + lane * 4;
    unsafeAtomicAdd(dst + 0, w * hv.x);
    unsafeAtomicAdd(dst + 1, w * hv.y);
    unsafeAtomicAdd(dst + 2, w * hv.z);
    unsafeAtomicAdd(dst + 3, w * hv.w);
}

// ----------------------------------- bias + relu + fp32 -> f16 convert ------
__global__ void k_act(const float* __restrict__ agg, const float* __restrict__ b,
                      _Float16* __restrict__ h16) {
    int t = blockIdx.x * blockDim.x + threadIdx.x;
    if (t < N_NODES * N_HID) {
        float v = agg[t] + b[t & 127];
        v = v > 0.0f ? v : 0.0f;
        h16[t] = (_Float16)v;
    }
}

// ------------------------------------------- out = H H^T via WMMA f16 -------
// block = 256 threads (8 waves), 128x128 output tile, K=128 in 4 steps of 32.
// LDS rows padded to 136 halves (272B) to avoid 64-bank row conflicts.
#define LSTR 136
__global__ __launch_bounds__(256) void k_gemm(const _Float16* __restrict__ H16,
                                              float* __restrict__ out) {
    __shared__ __align__(16) _Float16 As[128 * LSTR];
    __shared__ __align__(16) _Float16 Bs[128 * LSTR];

    const int i0 = blockIdx.y * 128;
    const int j0 = blockIdx.x * 128;

    // cooperative tile staging: 128 rows x 16 uint4 segments per matrix,
    // via GLOBAL_LOAD_ASYNC_TO_LDS_B128 (no VGPR round-trip) when available
    for (int t = threadIdx.x; t < 128 * 16; t += 256) {
        int row = t >> 4, seg = t & 15;
        copy16_g2l(&As[row * LSTR + seg * 8],
                   ((const uint4*)(H16 + (size_t)(i0 + row) * N_HID)) + seg);
        copy16_g2l(&Bs[row * LSTR + seg * 8],
                   ((const uint4*)(H16 + (size_t)(j0 + row) * N_HID)) + seg);
    }
    async_copies_wait();
    __syncthreads();

    const int lane = threadIdx.x & 31;
    const int wave = threadIdx.x >> 5;
    const int wr   = (wave >> 1) * 32;   // 0,32,64,96
    const int wc   = (wave & 1) * 64;    // 0,64
    const int mn   = lane & 15;
    const int ksel = lane >> 4;

    v8f acc[2][4] = {};

#pragma unroll
    for (int ks = 0; ks < 4; ++ks) {
        const int kbase = ks * 32;
        FragH af[2], bf[4];
#pragma unroll
        for (int ti = 0; ti < 2; ++ti) {
            const _Float16* s = &As[(wr + ti * 16 + mn) * LSTR + kbase + ksel * 8];
            af[ti].q[0] = *(const uint4*)(s);        // K = kbase + ksel*8      .. +7
            af[ti].q[1] = *(const uint4*)(s + 16);   // K = kbase + 16 + ksel*8 .. +7
        }
#pragma unroll
        for (int tj = 0; tj < 4; ++tj) {
            const _Float16* s = &Bs[(wc + tj * 16 + mn) * LSTR + kbase + ksel * 8];
            bf[tj].q[0] = *(const uint4*)(s);        // K = kbase + ksel*8      .. +7
            bf[tj].q[1] = *(const uint4*)(s + 16);   // K = kbase + 16 + ksel*8 .. +7
        }
#pragma unroll
        for (int ti = 0; ti < 2; ++ti)
#pragma unroll
            for (int tj = 0; tj < 4; ++tj)
                acc[ti][tj] = __builtin_amdgcn_wmma_f32_16x16x32_f16(
                    false, af[ti].v, false, bf[tj].v,
                    (short)0, acc[ti][tj], false, false);
    }

    // D layout: VGPR r -> M = r + 8*ksel, N = lane&15
#pragma unroll
    for (int ti = 0; ti < 2; ++ti)
#pragma unroll
        for (int tj = 0; tj < 4; ++tj)
#pragma unroll
            for (int r = 0; r < 8; ++r) {
                size_t row = (size_t)(i0 + wr + ti * 16 + r + 8 * ksel);
                out[row * N_NODES + (j0 + wc + tj * 16 + mn)] = acc[ti][tj][r];
            }
}

// ---------------------------------------------------------------------------
extern "C" void kernel_launch(void* const* d_in, const int* in_sizes, int n_in,
                              void* d_out, int out_size, void* d_ws, size_t ws_size,
                              hipStream_t stream) {
    const float* x  = (const float*)d_in[0];
    const int*   ei = (const int*)d_in[1];
    const float* W  = (const float*)d_in[2];
    const float* b  = (const float*)d_in[3];
    float* out = (float*)d_out;

    char* ws = (char*)d_ws;
    float*    deg  = (float*)(ws);                                  // 64 KB
    float*    dinv = (float*)(ws + 65536);                          // 64 KB
    float*    h    = (float*)(ws + 131072);                         // 8 MB
    float*    agg  = (float*)(ws + 131072 + 8388608);               // 8 MB
    _Float16* h16  = (_Float16*)(ws + 131072 + 2 * 8388608);        // 4 MB

    k_init_deg<<<N_NODES / 256, 256, 0, stream>>>(deg);
    k_edge_deg<<<N_EDGES / 256, 256, 0, stream>>>(ei, deg);
    k_dinv<<<N_NODES / 256, 256, 0, stream>>>(deg, dinv);
    k_xw<<<N_NODES / 32, 256, 0, stream>>>(x, W, h);
    k_self<<<(N_NODES * N_HID) / 256, 256, 0, stream>>>(h, dinv, agg);
    k_scatter<<<N_EDGES / 8, 256, 0, stream>>>(ei, h, dinv, agg);
    k_act<<<(N_NODES * N_HID) / 256, 256, 0, stream>>>(agg, b, h16);

    dim3 grid(N_NODES / 128, N_NODES / 128);
    k_gemm<<<grid, 256, 0, stream>>>(h16, out);
}
